// BahdanauAttention_73658689126629
// MI455X (gfx1250) — matrix operand order repaired and checked by reading, not compiled, not run on hardware
//
#include <hip/hip_runtime.h>
#include <hip/hip_bf16.h>

// ---------------------------------------------------------------------------
// Bahdanau attention, MI455X (gfx1250), wave32 + v_wmma_f32_16x16x32_bf16
// + GLOBAL_LOAD_ASYNC_TO_LDS staging.   B=32, T=4096, D=512, U=256.
// ---------------------------------------------------------------------------

#define BB 32
#define TT 4096
#define DD 512
#define UU 256
#define ROWS 32              // T-rows per block (2 M-tiles of 16)

typedef __attribute__((ext_vector_type(16))) __bf16 v16bf;
typedef __attribute__((ext_vector_type(8)))  __bf16 v8bf;
typedef __attribute__((ext_vector_type(8)))  float  v8f;
typedef int v4i __attribute__((vector_size(16)));
typedef __attribute__((address_space(1))) v4i* gvec_p;   // global 16B vector
typedef __attribute__((address_space(3))) v4i* lvec_p;   // LDS    16B vector

#if defined(__has_builtin)
#  if __has_builtin(__builtin_amdgcn_global_load_async_to_lds_b128)
#    define HAVE_ASYNC_LDS 1
#  endif
#  if __has_builtin(__builtin_amdgcn_s_wait_asynccnt)
#    define HAVE_WAIT_ASYNC 1
#  endif
#endif

__device__ __forceinline__ __bf16 f2bf(float f) {
    unsigned u = __builtin_bit_cast(unsigned, f);
    u += 0x7FFFu + ((u >> 16) & 1u);               // round-to-nearest-even
    unsigned short h = (unsigned short)(u >> 16);
    return __builtin_bit_cast(__bf16, h);
}

// ---------------------------------------------------------------------------
// Kernel 1: hq[b,u] = query[b,:]·W2[:,u] + b2[u]; also zero context output.
// grid = (B), block = (U)
// ---------------------------------------------------------------------------
__global__ void prep_hq_kernel(const float* __restrict__ query,
                               const float* __restrict__ W2,
                               const float* __restrict__ b2,
                               float* __restrict__ hq,
                               float* __restrict__ ctx) {
    const int b = blockIdx.x;
    const int u = threadIdx.x;
    float acc = b2[u];
    const float* q = query + b * DD;
    #pragma unroll 4
    for (int d = 0; d < DD; ++d)
        acc = fmaf(q[d], W2[d * UU + u], acc);
    hq[b * UU + u] = acc;
    ctx[b * DD + u]      = 0.0f;
    ctx[b * DD + UU + u] = 0.0f;
}

// ---------------------------------------------------------------------------
// Kernel 2: W1T[n][k] = bf16(W1[k][n])   (N-major for WMMA B-fragment loads)
// grid = (U), block = (256)
// ---------------------------------------------------------------------------
__global__ void w1_transpose_kernel(const float* __restrict__ W1,
                                    __bf16* __restrict__ W1T) {
    const int n = blockIdx.x;
    for (int k = threadIdx.x; k < DD; k += blockDim.x)
        W1T[n * DD + k] = f2bf(W1[k * UU + n]);
}

// ---------------------------------------------------------------------------
// Kernel 3: fused score GEMM.
//   score[b,t] = tanh(values[b,t,:]@W1 + b1 + hq[b,:]) · V + bV
// One block per (b, 32-row T tile). 8 waves; wave owns 2 N-tiles of 16 and
// computes both 16-row M-tiles per N-tile (B-fragment register reuse).
// grid = (B * T/32), block = (256)
// ---------------------------------------------------------------------------
#define APITCH (DD + 8)   // bf16 elems per LDS row; pad to stagger banks

__global__ void __launch_bounds__(256)
score_gemm_kernel(const float* __restrict__ values,
                  const __bf16* __restrict__ W1T,
                  const float* __restrict__ b1,
                  const float* __restrict__ hq,
                  const float* __restrict__ Vw,
                  const float* __restrict__ bV,
                  float* __restrict__ scores) {
    __shared__ __align__(16) float  Stage[ROWS * DD];     // raw f32 tile (64KB)
    __shared__ __align__(16) __bf16 As[ROWS * APITCH];    // bf16 A buffer
    __shared__ float s_score[ROWS];

    const int tid  = threadIdx.x;
    const int b    = blockIdx.x >> 7;          // T/32 = 128 tiles per batch
    const int tile = blockIdx.x & 127;
    const int t0   = tile * ROWS;

    if (tid < ROWS) s_score[tid] = 0.0f;

    const float* gsrc = values + ((size_t)b * TT + t0) * DD;  // contiguous tile

#if defined(HAVE_ASYNC_LDS)
    // ---- async DMA: global f32 tile -> LDS staging (no VGPR data) ----
    #pragma unroll 4
    for (int i = 0; i < 16; ++i) {
        int c = tid + i * 256;                 // 4096 x 16B chunks
        __builtin_amdgcn_global_load_async_to_lds_b128(
            (gvec_p)(gsrc + c * 4),
            (lvec_p)(Stage + c * 4),
            0, 0);
    }
#  if defined(HAVE_WAIT_ASYNC)
    __builtin_amdgcn_s_wait_asynccnt(0);
#  else
    asm volatile("s_wait_asynccnt 0x0" ::: "memory");
#  endif
    __syncthreads();
    // ---- LDS f32 -> LDS bf16 (A-fragment row layout) ----
    #pragma unroll 4
    for (int i = 0; i < 16; ++i) {
        int idx4 = tid + i * 256;
        int row  = idx4 >> 7;                  // 128 float4 per row
        int c4   = idx4 & 127;
        float4 v = *(const float4*)(Stage + (row * DD + c4 * 4));
        __bf16* dst = &As[row * APITCH + c4 * 4];
        dst[0] = f2bf(v.x); dst[1] = f2bf(v.y);
        dst[2] = f2bf(v.z); dst[3] = f2bf(v.w);
    }
#else
    // ---- fallback: plain global loads -> convert -> LDS ----
    #pragma unroll 4
    for (int i = 0; i < 16; ++i) {
        int idx4 = tid + i * 256;
        int row  = idx4 >> 7;
        int c4   = idx4 & 127;
        float4 v = ((const float4*)gsrc)[row * 128 + c4];
        __bf16* dst = &As[row * APITCH + c4 * 4];
        dst[0] = f2bf(v.x); dst[1] = f2bf(v.y);
        dst[2] = f2bf(v.z); dst[3] = f2bf(v.w);
    }
#endif
    __syncthreads();

    const int wave = tid >> 5;
    const int lane = tid & 31;
    const int half = lane >> 4;                // 0: lanes 0-15, 1: lanes 16-31
    const int lrow = lane & 15;

    // A-fragment row bases for the two M-tiles (M = lrow, lrow+16)
    const __bf16* arow0 = &As[lrow * APITCH + 16 * half];
    const __bf16* arow1 = &As[(16 + lrow) * APITCH + 16 * half];

    for (int sub = 0; sub < 2; ++sub) {
        const int n0 = (wave * 2 + sub) * 16;  // N-tile base
        const int n  = n0 + lrow;              // this lane's N (C-layout)
        const __bf16* brow = W1T + (size_t)n * DD + 16 * half;

        v8f c0 = {}, c1 = {};
        #pragma unroll 4
        for (int k0 = 0; k0 < DD; k0 += 32) {
            // B: 32x16 bf16, N = lrow, K = k0 + 16*half .. +15 (contiguous)
            v16bf bm = *(const v16bf*)(brow + k0);
            // A: 16x32 bf16, K = k0 + 16*half .. +15 (contiguous)
            v8bf a0l = *(const v8bf*)(arow0 + k0);
            v8bf a0h = *(const v8bf*)(arow0 + k0 + 8);
            v16bf a0 = __builtin_shufflevector(a0l, a0h,
                         0,1,2,3,4,5,6,7,8,9,10,11,12,13,14,15);
            v8bf a1l = *(const v8bf*)(arow1 + k0);
            v8bf a1h = *(const v8bf*)(arow1 + k0 + 8);
            v16bf a1 = __builtin_shufflevector(a1l, a1h,
                         0,1,2,3,4,5,6,7,8,9,10,11,12,13,14,15);
            c0 = __builtin_amdgcn_wmma_f32_16x16x32_bf16(
                     false, a0, false, bm, (short)0, c0, false, false);
            c1 = __builtin_amdgcn_wmma_f32_16x16x32_bf16(
                     false, a1, false, bm, (short)0, c1, false, false);
        }

        // epilogue: tanh(h + b1 + hq) * V[n], reduce over N (16-lane halves)
        const float addn = b1[n] + hq[b * UU + n];
        const float vn   = Vw[n];
        #pragma unroll
        for (int r = 0; r < 8; ++r) {
            float t = tanhf(c0[r] + addn) * vn;
            t += __shfl_xor(t, 1);
            t += __shfl_xor(t, 2);
            t += __shfl_xor(t, 4);
            t += __shfl_xor(t, 8);
            if (lrow == 0) atomicAdd(&s_score[r + 8 * half], t);
            float u = tanhf(c1[r] + addn) * vn;
            u += __shfl_xor(u, 1);
            u += __shfl_xor(u, 2);
            u += __shfl_xor(u, 4);
            u += __shfl_xor(u, 8);
            if (lrow == 0) atomicAdd(&s_score[16 + r + 8 * half], u);
        }
    }
    __syncthreads();
    if (tid < ROWS)
        scores[(size_t)b * TT + t0 + tid] = s_score[tid] + bV[0];
}

// ---------------------------------------------------------------------------
// Kernel 4: in-place softmax over T per batch. grid=(B), block=(256)
// ---------------------------------------------------------------------------
__global__ void softmax_kernel(float* __restrict__ sc) {
    __shared__ float red[256];
    const int b   = blockIdx.x;
    const int tid = threadIdx.x;
    float* s = sc + (size_t)b * TT;

    float m = -3.4e38f;
    for (int t = tid; t < TT; t += 256) m = fmaxf(m, s[t]);
    red[tid] = m; __syncthreads();
    for (int st = 128; st > 0; st >>= 1) {
        if (tid < st) red[tid] = fmaxf(red[tid], red[tid + st]);
        __syncthreads();
    }
    const float bm = red[0];
    __syncthreads();

    float sum = 0.0f;
    for (int t = tid; t < TT; t += 256) sum += __expf(s[t] - bm);
    red[tid] = sum; __syncthreads();
    for (int st = 128; st > 0; st >>= 1) {
        if (tid < st) red[tid] += red[tid + st];
        __syncthreads();
    }
    const float inv = 1.0f / red[0];
    for (int t = tid; t < TT; t += 256) s[t] = __expf(s[t] - bm) * inv;
}

// ---------------------------------------------------------------------------
// Kernel 5: context[b,d] = sum_t attn[b,t] * values[b,t,d]
// grid = (B*16) (16 T-chunks of 256 per batch), block = (256); f32 atomics.
// ---------------------------------------------------------------------------
__global__ void __launch_bounds__(256)
context_kernel(const float* __restrict__ values,
               const float* __restrict__ attn,
               float* __restrict__ ctx) {
    const int b     = blockIdx.x >> 4;
    const int chunk = blockIdx.x & 15;
    const int t0    = chunk * 256;
    const int tid   = threadIdx.x;

    const float2* vrow = (const float2*)(values + ((size_t)b * TT + t0) * DD);
    const float*  a    = attn + (size_t)b * TT + t0;

    float2 acc = make_float2(0.0f, 0.0f);
    for (int i = 0; i < 256; ++i) {
        __builtin_prefetch(&vrow[(i + 6) * (DD / 2) + tid], 0, 0);
        float  w = a[i];
        float2 v = vrow[i * (DD / 2) + tid];
        acc.x = fmaf(w, v.x, acc.x);
        acc.y = fmaf(w, v.y, acc.y);
    }
    atomicAdd(&ctx[b * DD + 2 * tid],     acc.x);
    atomicAdd(&ctx[b * DD + 2 * tid + 1], acc.y);
}

// ---------------------------------------------------------------------------
extern "C" void kernel_launch(void* const* d_in, const int* in_sizes, int n_in,
                              void* d_out, int out_size, void* d_ws, size_t ws_size,
                              hipStream_t stream) {
    (void)in_sizes; (void)n_in; (void)out_size; (void)ws_size;
    const float* values = (const float*)d_in[0];   // [B,T,D]
    const float* query  = (const float*)d_in[1];   // [B,D]
    const float* W1     = (const float*)d_in[2];   // [D,U]
    const float* b1     = (const float*)d_in[3];   // [U]
    const float* W2     = (const float*)d_in[4];   // [D,U]
    const float* b2     = (const float*)d_in[5];   // [U]
    const float* Vw     = (const float*)d_in[6];   // [U,1]
    const float* bV     = (const float*)d_in[7];   // [1]

    float* out  = (float*)d_out;
    float* ctx  = out;                 // [B,D]   = 16384 floats
    float* attn = out + BB * DD;       // [B,T,1] = 131072 floats

    float*  hq  = (float*)d_ws;
    __bf16* W1T = (__bf16*)((char*)d_ws + BB * UU * sizeof(float));

    prep_hq_kernel      <<<BB,          UU,  0, stream>>>(query, W2, b2, hq, ctx);
    w1_transpose_kernel <<<UU,          256, 0, stream>>>(W1, W1T);
    score_gemm_kernel   <<<BB*(TT/ROWS), 256, 0, stream>>>(values, W1T, b1, hq,
                                                           Vw, bV, attn);
    softmax_kernel      <<<BB,          256, 0, stream>>>(attn);
    context_kernel      <<<BB*16,       256, 0, stream>>>(values, attn, ctx);
}
